// RelationPredictor_54082228191978
// MI455X (gfx1250) — matrix operand order repaired and checked by reading, not compiled, not run on hardware
//
#include <hip/hip_runtime.h>
#include <stdint.h>

#define NNODES   10000
#define NREL     50
#define RAUG     (2*NREL + 1)          // 101
#define DIM      128
#define NEDGES   320000
#define EAUG     (2*NEDGES + NNODES)   // 650000
#define NBATCH   65536
#define NSEG     (RAUG * NNODES)       // 1,010,000
#define NTILES   (NNODES / 16)         // 625 (exact)

#define GEMM_WAVES   8
#define GEMM_THREADS (GEMM_WAVES * 32)
#define STAGE_ELEMS  (4 * 8 * 512)               // bf16 elems of B per relation
#define STAGE_BYTES  (STAGE_ELEMS * 2)           // 32 KB
#define STAGE_CHUNKS (STAGE_BYTES / 16)          // 2048 x b128
#define CHUNKS_PER_T (STAGE_CHUNKS / GEMM_THREADS) // 8

typedef __attribute__((ext_vector_type(16))) __bf16 v16bf;
typedef __attribute__((ext_vector_type(8)))  float  v8f;

union ABfrag { uint4 u[2]; v16bf v; };

// ---------- helpers ----------

static __device__ __forceinline__ uint16_t f32_to_bf16(float f) {
  uint32_t u = __float_as_uint(f);
  return (uint16_t)((u + 0x7FFFu + ((u >> 16) & 1u)) >> 16);   // RNE
}

static __device__ __forceinline__ uint32_t pack_bf16x2(float lo, float hi) {
  uint32_t a = __float_as_uint(lo);
  uint32_t b = __float_as_uint(hi);
  a = (a + 0x7FFFu + ((a >> 16) & 1u)) >> 16;
  b = (b + 0x7FFFu + ((b >> 16) & 1u)) >> 16;
  return (a & 0xFFFFu) | (b << 16);
}

// Implicit edge augmentation: e in [0, EAUG)
static __device__ __forceinline__ void edge_decode(const int* __restrict__ g, int e,
                                                   int& src, int& rel, int& dst) {
  if (e < NEDGES) {
    src = g[3*e + 0]; rel = g[3*e + 1]; dst = g[3*e + 2];
  } else if (e < 2*NEDGES) {
    int e2 = e - NEDGES;
    src = g[3*e2 + 2]; rel = g[3*e2 + 1] + NREL; dst = g[3*e2 + 0];
  } else {
    int n = e - 2*NEDGES;
    src = n; rel = 2*NREL; dst = n;
  }
}

// ---------- kernels ----------

__global__ __launch_bounds__(256) void zero_u4(uint4* __restrict__ p, size_t n16) {
  size_t i      = (size_t)blockIdx.x * blockDim.x + threadIdx.x;
  size_t stride = (size_t)gridDim.x * blockDim.x;
  const uint4 z = {0u, 0u, 0u, 0u};
  for (; i < n16; i += stride) p[i] = z;
}

__global__ __launch_bounds__(256) void rgcn_deg(const int* __restrict__ g,
                                                int* __restrict__ deg) {
  const int e = blockIdx.x * blockDim.x + threadIdx.x;
  if (e >= EAUG) return;
  int src, rel, dst;
  edge_decode(g, e, src, rel, dst);
  atomicAdd(&deg[rel * NNODES + dst], 1);
}

// Pack W[r,d,o] (f32) into per-WMMA-fragment bf16 layout:
// fragment (r, kc, ot): lane L holds B[k, col] for col = ot*16 + (L&15),
// k = kc*32 + (L>>4)*16 + t, t = 0..15, stored contiguously (32B per lane).
__global__ __launch_bounds__(256) void rgcn_packW(const float* __restrict__ W,
                                                  uint16_t* __restrict__ wp) {
  const int idx = blockIdx.x * blockDim.x + threadIdx.x;
  if (idx >= RAUG * DIM * DIM) return;
  const int t    = idx & 15;
  const int lane = (idx >> 4) & 31;
  const int frag = idx >> 9;
  const int ot   = frag & 7;
  const int kc   = (frag >> 3) & 3;
  const int r    = frag >> 5;
  const int k    = kc * 32 + ((lane >> 4) * 16) + t;
  const int col  = ot * 16 + (lane & 15);
  wp[idx] = f32_to_bf16(W[((size_t)r * DIM + k) * DIM + col]);
}

// Scatter normalized messages into bf16 agg with native packed-bf16 atomics.
__global__ __launch_bounds__(256) void rgcn_scatter(const int* __restrict__ g,
                                                    const float* __restrict__ x,
                                                    const int* __restrict__ deg,
                                                    uint16_t* __restrict__ agg) {
  const long long t = (long long)blockIdx.x * blockDim.x + threadIdx.x;
  if (t >= (long long)EAUG * 64) return;
  const int e  = (int)(t >> 6);
  const int pr = (int)(t & 63);
  int src, rel, dst;
  edge_decode(g, e, src, rel, dst);
  const int   seg  = rel * NNODES + dst;
  const int   dgi  = deg[seg];
  const float norm = 1.0f / (float)(dgi > 1 ? dgi : 1);
  const float2 xv  = *(const float2*)(x + (size_t)src * DIM + pr * 2);
  const uint32_t pk = pack_bf16x2(xv.x * norm, xv.y * norm);
  uint16_t* addr = agg + ((size_t)seg * DIM + pr * 2);
  asm volatile("global_atomic_pk_add_bf16 %0, %1, off"
               :: "v"((uint64_t)(uintptr_t)addr), "v"(pk) : "memory");
}

// ---- GEMM: out[n,o] = act( sum_r agg[r,n,:] @ W[r,:,o] + bias[o] ) ----
// 8 waves/block, one 16-row node tile per wave. B fragments for relation r
// (32 KB) are staged in LDS via async global->LDS copies, double-buffered,
// and shared by all 8 waves. A fragments are double-buffered in registers.
// B-fragment LDS reads are software-pipelined 2 deep against the WMMAs.

static __device__ __forceinline__ void gemm_issue_stage(const uint16_t* __restrict__ wpack,
                                                        int r, uint32_t lds_base, int tid) {
  const char* gsrc = (const char*)(wpack + (size_t)r * STAGE_ELEMS);
#pragma unroll
  for (int i = 0; i < CHUNKS_PER_T; ++i) {
    const int c = tid + i * GEMM_THREADS;
    const uint64_t ga = (uint64_t)(uintptr_t)(gsrc + (size_t)c * 16);
    const uint32_t la = lds_base + (uint32_t)c * 16u;
    asm volatile("global_load_async_to_lds_b128 %0, %1, off"
                 :: "v"(la), "v"(ga) : "memory");
  }
}

static __device__ __forceinline__ void gemm_load_A(const uint16_t* __restrict__ agg,
                                                   int r, int row, int lo,
                                                   ABfrag (&a)[4]) {
  const uint16_t* arow = agg + ((size_t)r * NNODES + row) * DIM;
#pragma unroll
  for (int kc = 0; kc < 4; ++kc) {
    a[kc].u[0] = *(const uint4*)(arow + kc * 32 + lo);
    a[kc].u[1] = *(const uint4*)(arow + kc * 32 + 8 + lo);
  }
}

// 32 fragments per stage, fragment i at element offset i*512; double-buffered
// B registers so each WMMA overlaps with the next fragment's ds_load_b128.
static __device__ __forceinline__ void gemm_compute(const ABfrag (&a)[4],
                                                    const uint16_t* __restrict__ sb,
                                                    int lane, v8f (&acc)[8]) {
  const uint16_t* bbase = sb + (size_t)lane * 16;
  ABfrag b[2];
  {
    const uint4* bp = (const uint4*)bbase;
    b[0].u[0] = bp[0];
    b[0].u[1] = bp[1];
  }
#pragma unroll
  for (int i = 0; i < 32; ++i) {
    const int kc = i >> 3;
    const int ot = i & 7;
    if (i + 1 < 32) {
      const uint4* bp = (const uint4*)(bbase + (size_t)(i + 1) * 512);
      b[(i + 1) & 1].u[0] = bp[0];
      b[(i + 1) & 1].u[1] = bp[1];
    }
    acc[ot] = __builtin_amdgcn_wmma_f32_16x16x32_bf16(
        false, a[kc].v, false, b[i & 1].v, (short)0, acc[ot], false, false);
  }
}

#define ASYNC_WAIT_NEXT() asm volatile("s_wait_asynccnt 0x8" ::: "memory")
#define ASYNC_WAIT_ALL()  asm volatile("s_wait_asynccnt 0x0" ::: "memory")

__global__ __launch_bounds__(GEMM_THREADS) void rgcn_gemm(
    const uint16_t* __restrict__ agg, const uint16_t* __restrict__ wpack,
    const float* __restrict__ bias, float* __restrict__ out, int do_relu) {
  __shared__ uint8_t smem[2 * STAGE_BYTES];   // 64 KB double buffer

  const int tid   = threadIdx.x;
  const int lane  = tid & 31;
  const int wave  = tid >> 5;
  const int tile  = blockIdx.x * GEMM_WAVES + wave;
  const int live  = (tile < NTILES);
  const int n0    = (live ? tile : NTILES - 1) * 16;  // clamp: keep EXEC full, loads in-bounds
  const int half  = lane >> 4;                        // 0 | 1
  const int lo    = half << 4;                        // A-layout K sub-offset 0 | 16
  const int row   = n0 + (lane & 15);

  const uint32_t lds0 = (uint32_t)(uintptr_t)&smem[0];
  const uint32_t lds1 = (uint32_t)(uintptr_t)&smem[STAGE_BYTES];
  const uint16_t* sb0 = (const uint16_t*)&smem[0];
  const uint16_t* sb1 = (const uint16_t*)&smem[STAGE_BYTES];

  v8f acc[8];
#pragma unroll
  for (int i = 0; i < 8; ++i)
    acc[i] = (v8f){0.f, 0.f, 0.f, 0.f, 0.f, 0.f, 0.f, 0.f};

  ABfrag A0[4], A1[4];
  gemm_issue_stage(wpack, 0, lds0, tid);
  gemm_load_A(agg, 0, row, lo, A0);

  for (int r = 0; r < RAUG; r += 2) {
    // ---- even stage r: compute from (A0, sb0), prefetch r+1 -> (A1, sb1) ----
    if (r + 1 < RAUG) {
      gemm_issue_stage(wpack, r + 1, lds1, tid);
      gemm_load_A(agg, r + 1, row, lo, A1);
      ASYNC_WAIT_NEXT();         // stage r complete (stage r+1 still in flight)
    } else {
      ASYNC_WAIT_ALL();
    }
    __syncthreads();             // all waves' async portions of stage r visible
    gemm_compute(A0, sb0, lane, acc);
    __syncthreads();             // buffer 0 free for reuse

    // ---- odd stage r+1: compute from (A1, sb1), prefetch r+2 -> (A0, sb0) ----
    if (r + 1 < RAUG) {
      if (r + 2 < RAUG) {
        gemm_issue_stage(wpack, r + 2, lds0, tid);
        gemm_load_A(agg, r + 2, row, lo, A0);
        ASYNC_WAIT_NEXT();
      } else {
        ASYNC_WAIT_ALL();
      }
      __syncthreads();
      gemm_compute(A1, sb1, lane, acc);
      __syncthreads();
    }
  }

  // C/D layout: lanes 0-15 -> M=m, lanes 16-31 -> M=m+8; N = lane&15
  if (live) {
    const int col = lane & 15;
#pragma unroll
    for (int ot = 0; ot < 8; ++ot) {
      const float bo = bias[ot * 16 + col];
#pragma unroll
      for (int m = 0; m < 8; ++m) {
        float v = acc[ot][m] + bo;
        if (do_relu) v = fmaxf(v, 0.0f);
        out[(size_t)(n0 + m + 8 * half) * DIM + ot * 16 + col] = v;
      }
    }
  }
}

// out[i] = sum_d h2[s,d] * rel[p,d] * h2[o,d] ; one wave per batch item
__global__ __launch_bounds__(256) void rgcn_score(const int* __restrict__ batch,
                                                  const float* __restrict__ h2,
                                                  const float* __restrict__ rels,
                                                  float* __restrict__ out) {
  const int i    = blockIdx.x * 8 + (threadIdx.x >> 5);
  const int lane = threadIdx.x & 31;
  if (i >= NBATCH) return;
  const int s = batch[3*i + 0];
  const int p = batch[3*i + 1];
  const int o = batch[3*i + 2];
  const float4 a = *(const float4*)(h2   + (size_t)s * DIM + lane * 4);
  const float4 r = *(const float4*)(rels + (size_t)p * DIM + lane * 4);
  const float4 c = *(const float4*)(h2   + (size_t)o * DIM + lane * 4);
  float acc = a.x*r.x*c.x + a.y*r.y*c.y + a.z*r.z*c.z + a.w*r.w*c.w;
#pragma unroll
  for (int off = 16; off > 0; off >>= 1) acc += __shfl_down(acc, off, 32);
  if (lane == 0) out[i] = acc;
}

// ---------- launch ----------

extern "C" void kernel_launch(void* const* d_in, const int* in_sizes, int n_in,
                              void* d_out, int out_size, void* d_ws, size_t ws_size,
                              hipStream_t stream) {
  (void)in_sizes; (void)n_in; (void)out_size; (void)ws_size;
  const int*   graph = (const int*)  d_in[0];
  const int*   batch = (const int*)  d_in[1];
  const float* emb   = (const float*)d_in[2];
  const float* W1    = (const float*)d_in[3];
  const float* b1    = (const float*)d_in[4];
  const float* W2    = (const float*)d_in[5];
  const float* b2    = (const float*)d_in[6];
  const float* rels  = (const float*)d_in[7];
  float* out = (float*)d_out;

  char* ws = (char*)d_ws;
  size_t off = 0;
  auto carve = [&](size_t bytes) -> void* {
    void* p = ws + off;
    off = (off + bytes + 255) & ~(size_t)255;
    return p;
  };
  int*      deg = (int*)     carve((size_t)NSEG * 4);
  uint16_t* agg = (uint16_t*)carve((size_t)NSEG * DIM * 2);       // 258.6 MB bf16
  float*    h1  = (float*)   carve((size_t)NNODES * DIM * 4);
  float*    h2  = (float*)   carve((size_t)NNODES * DIM * 4);
  uint16_t* wp1 = (uint16_t*)carve((size_t)RAUG * DIM * DIM * 2); // 3.3 MB
  uint16_t* wp2 = (uint16_t*)carve((size_t)RAUG * DIM * DIM * 2);

  const size_t deg16 = ((size_t)NSEG * 4) / 16;        // 252,500
  const size_t agg16 = ((size_t)NSEG * DIM * 2) / 16;  // 16,160,000
  const int    npk   = RAUG * DIM * DIM;               // 1,654,784
  const int    scat_blocks = (int)(((long long)EAUG * 64) / 256); // 162,500
  const int    gemm_blocks = (NTILES + GEMM_WAVES - 1) / GEMM_WAVES; // 79

  // degrees (shared by both layers)
  zero_u4<<<1024, 256, 0, stream>>>((uint4*)deg, deg16);
  rgcn_deg<<<(EAUG + 255) / 256, 256, 0, stream>>>(graph, deg);

  // weight packing (bf16, WMMA B-fragment layout)
  rgcn_packW<<<(npk + 255) / 256, 256, 0, stream>>>(W1, wp1);
  rgcn_packW<<<(npk + 255) / 256, 256, 0, stream>>>(W2, wp2);

  // layer 1
  zero_u4<<<8192, 256, 0, stream>>>((uint4*)agg, agg16);
  rgcn_scatter<<<scat_blocks, 256, 0, stream>>>(graph, emb, deg, agg);
  rgcn_gemm<<<gemm_blocks, GEMM_THREADS, 0, stream>>>(agg, wp1, b1, h1, 1);

  // layer 2
  zero_u4<<<8192, 256, 0, stream>>>((uint4*)agg, agg16);
  rgcn_scatter<<<scat_blocks, 256, 0, stream>>>(graph, h1, deg, agg);
  rgcn_gemm<<<gemm_blocks, GEMM_THREADS, 0, stream>>>(agg, wp2, b2, h2, 0);

  // scoring
  rgcn_score<<<NBATCH / 8, 256, 0, stream>>>(batch, h2, rels, out);
}
